// GGNNLayer_69784628625698
// MI455X (gfx1250) — matrix operand order repaired and checked by reading, not compiled, not run on hardware
//
#include <hip/hip_runtime.h>
#include <hip/hip_bf16.h>

typedef __attribute__((ext_vector_type(16))) _Float16 v16h;
typedef __attribute__((ext_vector_type(8)))  _Float16 v8h;
typedef __attribute__((ext_vector_type(8)))  float    v8f;

#define GG_N  50000
#define GG_E  800000
#define GG_D  128
#define GG_ED 6
#define GG_M  262     // 2*D + ED
#define KPAD  288     // M padded to multiple of 32 (9 k-tiles)
#define NT1   18      // layer-1 out tiles padded to 288 (uniform, no y pre-zero)
#define KT1   9
#define NT2   8       // layer-2 out tiles (128)
#define KT2   9       // layer-2 K = 262 padded to 288
#define GKT   4       // GRU K tiles (128)
#define FRAG_HALFS 512  // 32 lanes * 16 halves per B fragment

// ---------------------------------------------------------------------------
// Fragment layouts per CDNA5 WMMA 16x16x32 f16:
// A: lane l -> row = l%16, group = l/16; half j -> k = (j/8)*16 + group*8 + (j%8)
//    => two contiguous 8-half (16B) runs per fragment.
// B (packed): lane l -> col = l%16; half j -> k = (j/8)*16 + (l/16)*8 + (j%8); 16 halves
//    contiguous per lane => one 32B load.
// C/D: lane l, vgpr r -> row = r + 8*(l/16), col = l%16.
// ---------------------------------------------------------------------------
__device__ __forceinline__ v16h load_a_frag(const _Float16* __restrict__ tile,
                                            int stride, int row, int kt, int group) {
    const _Float16* base = tile + row * stride + kt * 32 + group * 8;
    union { v16h v; v8h h[2]; } u;
    u.h[0] = *(const v8h*)(base);
    u.h[1] = *(const v8h*)(base + 16);
    return u.v;
}

__device__ __forceinline__ v16h load_b_frag(const _Float16* __restrict__ packed,
                                            int fragIdx, int lane) {
    return *(const v16h*)(packed + (size_t)fragIdx * FRAG_HALFS + lane * 16);
}

__device__ __forceinline__ v8f wmma_f16(v16h a, v16h b, v8f c) {
    return __builtin_amdgcn_wmma_f32_16x16x32_f16(false, a, false, b,
                                                  (short)0, c, false, false);
}

// ---------------------------------------------------------------------------
__global__ void pack_b_kernel(const float* __restrict__ W, _Float16* __restrict__ dst,
                              int nout, int kin, int ntiles, int ktiles) {
    int total = ntiles * ktiles * FRAG_HALFS;
    for (int tid = blockIdx.x * blockDim.x + threadIdx.x; tid < total;
         tid += gridDim.x * blockDim.x) {
        int f    = tid >> 9;
        int rem  = tid & 511;
        int lane = rem >> 4;
        int j    = rem & 15;
        int kt   = f % ktiles;
        int nt   = f / ktiles;
        int n = nt * 16 + (lane & 15);
        int k = kt * 32 + ((j < 8) ? (lane >> 4) * 8 + j : 16 + (lane >> 4) * 8 + (j - 8));
        float v = (n < nout && k < kin) ? W[(size_t)n * kin + k] : 0.0f;
        dst[tid] = (_Float16)v;
    }
}

__global__ void zero_kernel(float* __restrict__ a_v, int* __restrict__ deg,
                            int navf, int ndeg) {
    int total = navf + ndeg;
    for (int i = blockIdx.x * blockDim.x + threadIdx.x; i < total;
         i += gridDim.x * blockDim.x) {
        if (i < navf) a_v[i] = 0.0f;
        else          deg[i - navf] = 0;
    }
}

// ---------------------------------------------------------------------------
// Edge MLP: ONE wave per block, 32 edges per wave as TWO 16-edge M-tiles.
// Each B fragment is loaded once and feeds two independent WMMA chains
// (2-way ILP on the matrix pipe + halved B-fragment L2 traffic).
// ---------------------------------------------------------------------------
__global__ void __launch_bounds__(32)
edge_mlp_kernel(const float* __restrict__ h, const float* __restrict__ e,
                const int* __restrict__ esrc, const int* __restrict__ edst,
                const _Float16* __restrict__ Wp1, const _Float16* __restrict__ Wp2,
                float* __restrict__ a_v, int* __restrict__ deg) {
    __shared__ __align__(32) _Float16 sM[2][16 * KPAD];
    __shared__ __align__(32) _Float16 sY[2][16 * KPAD];
    __shared__ int sDst[32];
    __shared__ int sSrc[32];

    const int lane = threadIdx.x;
    const int t0   = blockIdx.x * 32;          // E/32 = 25000 blocks, exact

    sDst[lane] = edst[t0 + lane];
    sSrc[lane] = esrc[t0 + lane];
    __syncthreads();

    // build both message tiles (f32 -> f16); lanes sweep consecutive cols
    for (int i = lane; i < 2 * 16 * KPAD; i += 32) {
        int t    = i / (16 * KPAD);
        int rem  = i - t * (16 * KPAD);
        int row  = rem / KPAD;
        int col  = rem - row * KPAD;
        int eidx = t * 16 + row;               // 0..31 within block
        float v = 0.0f;
        if (col < GG_D) {
            v = h[(size_t)sDst[eidx] * GG_D + col];
        } else if (col < 2 * GG_D) {
            v = h[(size_t)sSrc[eidx] * GG_D + (col - GG_D)];
        } else if (col < GG_M) {
            v = e[(size_t)(t0 + eidx) * GG_ED + (col - 2 * GG_D)];
        }
        sM[t][rem] = (_Float16)v;
    }
    __syncthreads();

    const int group = lane >> 4;
    const int cid   = lane & 15;

    // ---- layer 1: y = relu(m @ We1^T), 18 n-tiles x 9 k-tiles, dual M chains
    for (int nt = 0; nt < NT1; ++nt) {
        v8f acc0 = {}, acc1 = {};
        for (int kt = 0; kt < KT1; ++kt) {
            v16h b  = load_b_frag(Wp1, nt * KT1 + kt, lane);
            v16h a0 = load_a_frag(sM[0], KPAD, cid, kt, group);
            v16h a1 = load_a_frag(sM[1], KPAD, cid, kt, group);
            acc0 = wmma_f16(a0, b, acc0);
            acc1 = wmma_f16(a1, b, acc1);
        }
#pragma unroll
        for (int r = 0; r < 8; ++r) {
            int off = (r + 8 * group) * KPAD + nt * 16 + cid;
            float v0 = acc0[r]; v0 = v0 > 0.0f ? v0 : 0.0f;
            float v1 = acc1[r]; v1 = v1 > 0.0f ? v1 : 0.0f;
            sY[0][off] = (_Float16)v0;
            sY[1][off] = (_Float16)v1;
        }
    }
    __syncthreads();

    // ---- layer 2: a = y @ We2^T, 8 n-tiles x 9 k-tiles, dual M chains
    for (int nt = 0; nt < NT2; ++nt) {
        v8f acc0 = {}, acc1 = {};
        for (int kt = 0; kt < KT2; ++kt) {
            v16h b  = load_b_frag(Wp2, nt * KT2 + kt, lane);
            v16h a0 = load_a_frag(sY[0], KPAD, cid, kt, group);
            v16h a1 = load_a_frag(sY[1], KPAD, cid, kt, group);
            acc0 = wmma_f16(a0, b, acc0);
            acc1 = wmma_f16(a1, b, acc1);
        }
#pragma unroll
        for (int r = 0; r < 8; ++r) {
            int row = r + 8 * group;
            int col = nt * 16 + cid;
            unsafeAtomicAdd(&a_v[(size_t)sDst[row]      * GG_D + col], acc0[r]);
            unsafeAtomicAdd(&a_v[(size_t)sDst[16 + row] * GG_D + col], acc1[r]);
        }
    }

    atomicAdd(&deg[sDst[lane]], 1);
}

// ---------------------------------------------------------------------------
// Fused GRU + graph-norm + layer-norm + relu: one wave per 16 nodes.
// r/z accumulate both matmuls (a_v@Wih and h@Whh) into one WMMA chain each;
// four independent chains per n-tile keep the matrix pipe busy.
// ---------------------------------------------------------------------------
__global__ void __launch_bounds__(64)
gru_kernel(const float* __restrict__ h, const float* __restrict__ a_v,
           const int* __restrict__ deg, const float* __restrict__ snorm,
           const _Float16* __restrict__ Wih, const _Float16* __restrict__ Whh,
           const float* __restrict__ ln_g, const float* __restrict__ ln_b,
           float* __restrict__ out) {
    __shared__ __align__(32) _Float16 sA[2][16 * GG_D];
    __shared__ __align__(32) _Float16 sH[2][16 * GG_D];
    __shared__ float sHN[2][16 * GG_D];
    __shared__ float sMu[2][16];
    __shared__ float sRs[2][16];

    const int wave = threadIdx.x >> 5;
    const int lane = threadIdx.x & 31;
    int tile = blockIdx.x * 2 + wave;
    const int NTILES = GG_N / 16;            // 3125
    if (tile >= NTILES) tile = NTILES - 1;   // duplicate work; identical writes
    const int r0 = tile * 16;

    _Float16* aT  = sA[wave];
    _Float16* hT  = sH[wave];
    float*    hnT = sHN[wave];

    for (int i = lane; i < 16 * GG_D; i += 32) {
        int row = i >> 7;
        int col = i & 127;
        float hv = h[(size_t)(r0 + row) * GG_D + col];
        float av = (deg[r0 + row] > 0) ? a_v[(size_t)(r0 + row) * GG_D + col] : hv;
        aT[i] = (_Float16)av;
        hT[i] = (_Float16)hv;
    }
    __syncthreads();

    const int group = lane >> 4;
    const int cid   = lane & 15;

    for (int nt = 0; nt < 8; ++nt) {
        v8f ar = {}, az = {}, ain = {}, ahn = {};
        for (int kt = 0; kt < GKT; ++kt) {
            v16h aa = load_a_frag(aT, GG_D, cid, kt, group);
            v16h hh = load_a_frag(hT, GG_D, cid, kt, group);
            ar  = wmma_f16(aa, load_b_frag(Wih, (0 * 8 + nt) * GKT + kt, lane), ar);
            ar  = wmma_f16(hh, load_b_frag(Whh, (0 * 8 + nt) * GKT + kt, lane), ar);
            az  = wmma_f16(aa, load_b_frag(Wih, (1 * 8 + nt) * GKT + kt, lane), az);
            az  = wmma_f16(hh, load_b_frag(Whh, (1 * 8 + nt) * GKT + kt, lane), az);
            ain = wmma_f16(aa, load_b_frag(Wih, (2 * 8 + nt) * GKT + kt, lane), ain);
            ahn = wmma_f16(hh, load_b_frag(Whh, (2 * 8 + nt) * GKT + kt, lane), ahn);
        }
#pragma unroll
        for (int r = 0; r < 8; ++r) {
            int row = r + 8 * group;
            int gr  = r0 + row;
            int col = nt * 16 + cid;
            float rr = 1.0f / (1.0f + __expf(-ar[r]));
            float zz = 1.0f / (1.0f + __expf(-az[r]));
            float nn = tanhf(ain[r] + rr * ahn[r]);
            float hv = h[(size_t)gr * GG_D + col];
            float v  = (1.0f - zz) * nn + zz * hv;
            v *= snorm[gr];
            hnT[row * GG_D + col] = v;
        }
    }
    __syncthreads();

    if (lane < 16) {
        float s = 0.0f, s2 = 0.0f;
        for (int j = 0; j < GG_D; ++j) {
            float v = hnT[lane * GG_D + j];
            s += v; s2 += v * v;
        }
        float mu  = s * (1.0f / GG_D);
        float var = s2 * (1.0f / GG_D) - mu * mu;
        sMu[wave][lane] = mu;
        sRs[wave][lane] = rsqrtf(var + 1e-5f);
    }
    __syncthreads();

    for (int i = lane; i < 16 * GG_D; i += 32) {
        int row = i >> 7;
        int col = i & 127;
        float v = (hnT[i] - sMu[wave][row]) * sRs[wave][row] * ln_g[col] + ln_b[col];
        out[(size_t)(r0 + row) * GG_D + col] = fmaxf(v, 0.0f);
    }
}

// ---------------------------------------------------------------------------
extern "C" void kernel_launch(void* const* d_in, const int* in_sizes, int n_in,
                              void* d_out, int out_size, void* d_ws, size_t ws_size,
                              hipStream_t stream) {
    const float* h       = (const float*)d_in[0];
    const float* e       = (const float*)d_in[1];
    const float* snorm_n = (const float*)d_in[2];
    // d_in[3] = snorm_e (unused by reference output)
    const int*   esrc    = (const int*)d_in[4];
    const int*   edst    = (const int*)d_in[5];
    const float* W_e1    = (const float*)d_in[6];
    const float* W_e2    = (const float*)d_in[7];
    const float* W_ih    = (const float*)d_in[8];
    const float* W_hh    = (const float*)d_in[9];
    const float* ln_g    = (const float*)d_in[10];
    const float* ln_b    = (const float*)d_in[11];

    // workspace layout (f16 fragment-packed weights + a_v + deg), ~26.3 MB
    _Float16* Wp1  = (_Float16*)d_ws;
    _Float16* Wp2  = Wp1  + (size_t)NT1 * KT1 * FRAG_HALFS;   // 82944 halves
    _Float16* Wpih = Wp2  + (size_t)NT2 * KT2 * FRAG_HALFS;   // 36864 halves
    _Float16* Wphh = Wpih + (size_t)24 * GKT * FRAG_HALFS;    // 49152 halves
    float*    a_v  = (float*)(Wphh + (size_t)24 * GKT * FRAG_HALFS);
    int*      deg  = (int*)(a_v + (size_t)GG_N * GG_D);

    zero_kernel<<<2048, 256, 0, stream>>>(a_v, deg, GG_N * GG_D, GG_N);

    pack_b_kernel<<<340, 256, 0, stream>>>(W_e1, Wp1, GG_M, GG_M, NT1, KT1);
    pack_b_kernel<<<160, 256, 0, stream>>>(W_e2, Wp2, GG_D, GG_M, NT2, KT2);
    pack_b_kernel<<<192, 256, 0, stream>>>(W_ih, Wpih, 3 * GG_D, GG_D, 24, GKT);
    pack_b_kernel<<<192, 256, 0, stream>>>(W_hh, Wphh, 3 * GG_D, GG_D, 24, GKT);

    // 32 edges per single-wave block
    edge_mlp_kernel<<<GG_E / 32, 32, 0, stream>>>(h, e, esrc, edst, Wp1, Wp2, a_v, deg);

    // N/16 = 3125 tiles -> 1563 blocks (tail wave clamps, duplicate writes)
    gru_kernel<<<(GG_N / 16 + 1) / 2, 64, 0, stream>>>(h, a_v, deg, snorm_n,
                                                       Wpih, Wphh, ln_g, ln_b,
                                                       (float*)d_out);

    // second output: e passthrough
    hipMemcpyAsync((float*)d_out + (size_t)GG_N * GG_D, e,
                   (size_t)GG_E * GG_ED * sizeof(float),
                   hipMemcpyDeviceToDevice, stream);
}
